// SparseMLP_32676111188158
// MI455X (gfx1250) — matrix operand order repaired and checked by reading, not compiled, not run on hardware
//
#include <hip/hip_runtime.h>
#include <hip/hip_bf16.h>
#include <math.h>

// Problem constants (match reference)
#define BSZ 128
#define DD  1024
#define HH  131072
#define OO  1024
#define RR  512

// GEMM tiling
#define NBLK 128          // hidden rows per workgroup (8 waves x 16)
#define KCH  128          // K chunk staged in LDS
#define LPITCH 136        // bf16 row pitch in LDS: 272B -> +4 banks/row, conflict-free b128

typedef __attribute__((ext_vector_type(16))) __bf16         bf16x16;
typedef __attribute__((ext_vector_type(8)))  float          floatx8;
typedef __attribute__((ext_vector_type(8)))  unsigned int   uintx8;
typedef int v4i __attribute__((ext_vector_type(4)));

#if __has_builtin(__builtin_amdgcn_global_load_async_to_lds_b128)
#define HAVE_ASYNC_LDS 1
typedef __attribute__((address_space(1))) v4i gv4i;
typedef __attribute__((address_space(3))) v4i lv4i;
#else
#define HAVE_ASYNC_LDS 0
#endif

__device__ __forceinline__ unsigned int fkey(float f) {
    unsigned u = __float_as_uint(f);
    return (u & 0x80000000u) ? ~u : (u | 0x80000000u);   // monotonic orderable key
}

__device__ __forceinline__ float gelu_exact(float x) {
    return 0.5f * x * (1.0f + erff(x * 0.70710678118654752440f));
}

// A fragment (16x32 bf16): lane<16 holds K {k0..k0+15},
// lane>=16 holds K {k0+16..k0+31}; row M = lane&15 within the tile.
__device__ __forceinline__ bf16x16 load_a(const unsigned short* lds, int row, int k0, int hi) {
    const uint4* p0 = reinterpret_cast<const uint4*>(lds + row * LPITCH + k0 + (hi ? 16 : 0));
    const uint4* p1 = reinterpret_cast<const uint4*>(lds + row * LPITCH + k0 + 8 + (hi ? 16 : 0));
    uint4 a = *p0, b = *p1;
    uintx8 u = {a.x, a.y, a.z, a.w, b.x, b.y, b.z, b.w};
    return __builtin_bit_cast(bf16x16, u);
}

// B fragment (32x16 bf16): lane<16 -> N=lane, K=k..k+15; lane>=16 -> N=lane-16, K=k+16..k+31.
// Stream fp32 W_in (64B/lane), hardware RNE cvt (v_cvt_pk_bf16_f32).
__device__ __forceinline__ bf16x16 load_b(const float* __restrict__ wrow) {
    bf16x16 s;
#pragma unroll
    for (int i = 0; i < 16; i += 4) {
        float4 f = *reinterpret_cast<const float4*>(wrow + i);
        s[i]     = (__bf16)f.x;
        s[i + 1] = (__bf16)f.y;
        s[i + 2] = (__bf16)f.z;
        s[i + 3] = (__bf16)f.w;
    }
    return s;
}

__global__ __launch_bounds__(256) void k_cvt_x(const float* __restrict__ x,
                                               unsigned short* __restrict__ xb) {
    int i = blockIdx.x * 256 + threadIdx.x;
    if (i < BSZ * DD) xb[i] = __builtin_bit_cast(unsigned short, (__bf16)x[i]);
}

// scores[b, n] = sum_k x[b,k] * W_in[n,k]  via v_wmma_f32_16x16x32_bf16.
// Block: 256 thr = 8 waves; wave w owns N tile n0 = blk*128 + w*16, computes 8 M-tiles (M=128).
__global__ __launch_bounds__(256) void k_scores(const float* __restrict__ Win,
                                                const unsigned short* __restrict__ xb,
                                                float* __restrict__ scores) {
    __shared__ unsigned short lx[BSZ * LPITCH];   // 34,816 B
    const int tid  = threadIdx.x;
    const int wave = tid >> 5;
    const int lane = tid & 31;
    const int hi   = lane >> 4;
    const int lr   = lane & 15;
    const int n0   = blockIdx.x * NBLK + wave * 16;

    floatx8 acc[8];
#pragma unroll
    for (int m = 0; m < 8; ++m) acc[m] = (floatx8){0.f, 0.f, 0.f, 0.f, 0.f, 0.f, 0.f, 0.f};

    const unsigned int* xbu = reinterpret_cast<const unsigned int*>(xb);
    unsigned int*       lxu = reinterpret_cast<unsigned int*>(lx);
    const float*        wbase = Win + (size_t)(n0 + lr) * DD + hi * 16;

    for (int ch = 0; ch < DD / KCH; ++ch) {
        __syncthreads();
        // Stage x[:, ch*128 .. +128) as bf16 into LDS: 2048 x 16B granules, 8 per thread.
#pragma unroll
        for (int i = 0; i < (BSZ * KCH / 8) / 256; ++i) {   // 8 iters
            int t   = tid + i * 256;
            int row = t >> 4;        // 16 granules per row
            int c4  = t & 15;
            const unsigned int* src = xbu + row * (DD / 2) + ch * (KCH / 2) + c4 * 4;
            unsigned int*       dst = lxu + row * (LPITCH / 2) + c4 * 4;
#if HAVE_ASYNC_LDS
            __builtin_amdgcn_global_load_async_to_lds_b128((gv4i*)src, (lv4i*)dst, 0, 0);
#else
            *reinterpret_cast<uint4*>(dst) = *reinterpret_cast<const uint4*>(src);
#endif
        }
#if HAVE_ASYNC_LDS
        asm volatile("s_wait_asynccnt 0x0" ::: "memory");
#endif
        __syncthreads();
#pragma unroll
        for (int ks = 0; ks < KCH / 32; ++ks) {
            const int k0   = ks * 32;
            const int kabs = ch * KCH + k0;
            // keep the 512 MB W_in stream ahead: prefetch 2 chunks forward
            if (ch < DD / KCH - 2)
                __builtin_prefetch(wbase + kabs + 2 * KCH, 0, 3);
            bf16x16 bfrag = load_b(wbase + kabs);
            // software-pipeline A fragments one M-tile ahead (hide ds latency under WMMA)
            bf16x16 afrag = load_a(lx, lr, k0, hi);
#pragma unroll
            for (int m = 0; m < 8; ++m) {
                bf16x16 anext = (m < 7) ? load_a(lx, (m + 1) * 16 + lr, k0, hi) : afrag;
                acc[m] = __builtin_amdgcn_wmma_f32_16x16x32_bf16(
                    false, afrag, false, bfrag, (short)0, acc[m], false, false);
                afrag = anext;
            }
        }
    }

    // C layout: VGPR r, lanes 0-15 -> M = mt*16 + r, lanes 16-31 -> M = mt*16 + 8 + r; N = lane&15.
    const int col = n0 + lr;
#pragma unroll
    for (int m = 0; m < 8; ++m) {
#pragma unroll
        for (int r = 0; r < 8; ++r) {
            int brow = m * 16 + hi * 8 + r;
            scores[(size_t)brow * HH + col] = acc[m][r];
        }
    }
}

// Per-row exact top-512 via 3-level radix select on orderable keys (scores are L2-resident).
__global__ __launch_bounds__(1024) void k_select(const float* __restrict__ scores,
                                                 int* __restrict__ idx_out,
                                                 float* __restrict__ g_out) {
    __shared__ unsigned int hist[2048];
    __shared__ unsigned int sh[6];
    __shared__ unsigned int ctrG, ctrE;
    const int row = blockIdx.x;
    const int tid = threadIdx.x;
    const float* s = scores + (size_t)row * HH;

    // ---- level 1: bits [31:21]
    for (int i = tid; i < 2048; i += 1024) hist[i] = 0;
    __syncthreads();
    for (int i = tid; i < HH; i += 1024) atomicAdd(&hist[fkey(s[i]) >> 21], 1u);
    __syncthreads();
    if (tid == 0) {
        unsigned need = RR, cum = 0;
        for (int b = 2047;; --b) {
            unsigned c = hist[b];
            if (cum + c >= need) { sh[0] = (unsigned)b; sh[3] = need - cum; break; }
            cum += c;
        }
    }
    __syncthreads();
    const unsigned bin1 = sh[0], need1 = sh[3];

    // ---- level 2: bits [20:10] among matching prefix
    for (int i = tid; i < 2048; i += 1024) hist[i] = 0;
    __syncthreads();
    for (int i = tid; i < HH; i += 1024) {
        unsigned u = fkey(s[i]);
        if ((u >> 21) == bin1) atomicAdd(&hist[(u >> 10) & 0x7FFu], 1u);
    }
    __syncthreads();
    if (tid == 0) {
        unsigned need = need1, cum = 0;
        for (int b = 2047;; --b) {
            unsigned c = hist[b];
            if (cum + c >= need) { sh[1] = (unsigned)b; sh[4] = need - cum; break; }
            cum += c;
        }
    }
    __syncthreads();
    const unsigned bin2 = sh[1], need2 = sh[4];
    const unsigned pfx22 = (bin1 << 11) | bin2;

    // ---- level 3: bits [9:0] among matching prefix
    for (int i = tid; i < 2048; i += 1024) hist[i] = 0;
    __syncthreads();
    for (int i = tid; i < HH; i += 1024) {
        unsigned u = fkey(s[i]);
        if ((u >> 10) == pfx22) atomicAdd(&hist[u & 0x3FFu], 1u);
    }
    __syncthreads();
    if (tid == 0) {
        unsigned need = need2, cum = 0;
        for (int b = 1023;; --b) {
            unsigned c = hist[b];
            if (cum + c >= need) { sh[2] = (unsigned)b; sh[5] = need - cum; break; }
            cum += c;
        }
        ctrG = 0; ctrE = 0;
    }
    __syncthreads();
    const unsigned kstar  = (pfx22 << 10) | sh[2];
    const unsigned needEq = sh[5];
    const unsigned totG   = RR - needEq;

    // ---- compaction (coeff == score; gelu applied here)
    for (int i = tid; i < HH; i += 1024) {
        float f = s[i];
        unsigned u = fkey(f);
        if (u > kstar) {
            unsigned p = atomicAdd(&ctrG, 1u);
            idx_out[row * RR + p] = i;
            g_out[row * RR + p]   = gelu_exact(f);
        } else if (u == kstar) {
            unsigned e = atomicAdd(&ctrE, 1u);
            if (e < needEq) {
                unsigned p = totG + e;
                idx_out[row * RR + p] = i;
                g_out[row * RR + p]   = gelu_exact(f);
            }
        }
    }
}

// out[b,:] = sum_r g[b,r] * W_out[idx[b,r], :]   (streams gathered 4KB rows, float4 FMA)
__global__ __launch_bounds__(256) void k_out(const float* __restrict__ Wout,
                                             const int* __restrict__ idx,
                                             const float* __restrict__ g,
                                             float* __restrict__ out) {
    __shared__ int   sidx[RR];
    __shared__ float sg[RR];
    const int b = blockIdx.x, tid = threadIdx.x;
    for (int i = tid; i < RR; i += 256) { sidx[i] = idx[b * RR + i]; sg[i] = g[b * RR + i]; }
    __syncthreads();
    const int c0 = tid * 4;
    float4 acc = {0.f, 0.f, 0.f, 0.f};
    for (int r = 0; r < RR; ++r) {
        const float4 w = *reinterpret_cast<const float4*>(Wout + (size_t)sidx[r] * OO + c0);
        const float gg = sg[r];
        acc.x += gg * w.x; acc.y += gg * w.y; acc.z += gg * w.z; acc.w += gg * w.w;
    }
    *reinterpret_cast<float4*>(out + (size_t)b * OO + c0) = acc;
}

extern "C" void kernel_launch(void* const* d_in, const int* in_sizes, int n_in,
                              void* d_out, int out_size, void* d_ws, size_t ws_size,
                              hipStream_t stream) {
    const float* x    = (const float*)d_in[0];   // (128, 1024)
    const float* Win  = (const float*)d_in[1];   // (131072, 1024)
    const float* Wout = (const float*)d_in[2];   // (131072, 1024)
    // d_in[3] = sparsity_dim scalar (== 512, compiled in)

    char* w = (char*)d_ws;
    size_t off = 0;
    float* scores = (float*)(w + off);          off += (size_t)BSZ * HH * sizeof(float);  // 64 MB
    unsigned short* xb = (unsigned short*)(w + off); off += (size_t)BSZ * DD * 2;
    off = (off + 255) & ~(size_t)255;
    int* idx = (int*)(w + off);                 off += (size_t)BSZ * RR * sizeof(int);
    off = (off + 255) & ~(size_t)255;
    float* gco = (float*)(w + off);             off += (size_t)BSZ * RR * sizeof(float);

    k_cvt_x <<<(BSZ * DD) / 256, 256, 0, stream>>>(x, xb);
    k_scores<<<HH / NBLK,        256, 0, stream>>>(Win, xb, scores);
    k_select<<<BSZ,             1024, 0, stream>>>(scores, idx, gco);
    k_out   <<<BSZ,              256, 0, stream>>>(Wout, idx, gco, (float*)d_out);
}
